// FastWaveNet_69312182223119
// MI455X (gfx1250) — compile-verified, matrix operand order checked by reading, not run on hardware
//
#include <hip/hip_runtime.h>

// ---------------------------------------------------------------------------
// FastWaveNet autoregressive sampling for MI455X (gfx1250, wave32, WMMA, TDM).
// ---------------------------------------------------------------------------
#define RCH   256
#define SKIPC 256
#define CONDC 80
#define NCLS  256
#define TLEN  1024
#define NLAY  30      // dilations: (1<<(j%10)), j=0..29

typedef __attribute__((ext_vector_type(16))) __bf16        v16bf;
typedef __attribute__((ext_vector_type(8)))  float         v8f;
typedef __attribute__((ext_vector_type(4)))  float         v4f;
typedef __attribute__((ext_vector_type(8)))  unsigned int  v8u;
typedef __attribute__((ext_vector_type(4)))  unsigned int  v4u;
typedef __attribute__((ext_vector_type(4)))  unsigned int  u32x4;
typedef __attribute__((ext_vector_type(8)))  int           i32x8;
typedef __attribute__((ext_vector_type(4)))  int           i32x4;

// Tensor Data Mover availability (builtin arity differs between toolchains).
#if defined(__has_builtin)
#if __has_builtin(__builtin_amdgcn_tensor_load_to_lds) && \
    __has_builtin(__builtin_amdgcn_s_wait_tensorcnt)
#define HAVE_TDM 1
#endif
#endif
#ifndef HAVE_TDM
#define HAVE_TDM 0
#endif

// Device-pass-only probe: report in stderr whether the TDM path is compiled in.
#if defined(__AMDGCN__)
#if HAVE_TDM
#warning "CDNA5 TDM path ENABLED: tensor_load_to_lds + s_wait_tensorcnt in use"
#else
#warning "CDNA5 TDM builtins NOT available: falling back to global cond reads"
#endif
#endif

// ---------------------------------------------------------------------------
// Helpers
// ---------------------------------------------------------------------------
__device__ __forceinline__ unsigned int f2bf_pair(float lo, float hi) {
    unsigned int ulo = __builtin_bit_cast(unsigned int, lo);
    unsigned int uhi = __builtin_bit_cast(unsigned int, hi);
    ulo = ulo + 0x7FFFu + ((ulo >> 16) & 1u);
    uhi = uhi + 0x7FFFu + ((uhi >> 16) & 1u);
    return (ulo >> 16) | (uhi & 0xFFFF0000u);
}

// WMMA 16x16x32 bf16 per-lane K index for fragment dword p (ISA 7.12.2):
//   k = (p/4)*16 + (lane>=16 ? 8 : 0) + (p%4)*2   (pair k, k+1)
__device__ __forceinline__ int kidx(int lane, int p) {
    return ((p >> 2) << 4) + ((lane & 16) ? 8 : 0) + ((p & 3) << 1);
}

__device__ __forceinline__ v8f wmma_bf16(v8u a, v8u b, v8f c) {
    return __builtin_amdgcn_wmma_f32_16x16x32_bf16(
        false, __builtin_bit_cast(v16bf, a),
        false, __builtin_bit_cast(v16bf, b),
        (short)0, c, false, false);
}

__device__ __forceinline__ v8f vzero8() {
    v8f a;
#pragma unroll
    for (int i = 0; i < 8; ++i) a[i] = 0.f;
    return a;
}

// 8 consecutive floats via two b128 loads (32B-aligned pointers).
__device__ __forceinline__ v8f load8(const float* p) {
    v4f a = *(const v4f*)p;
    v4f b = *(const v4f*)(p + 4);
    v8f r;
#pragma unroll
    for (int i = 0; i < 4; ++i) { r[i] = a[i]; r[4 + i] = b[i]; }
    return r;
}
__device__ __forceinline__ void store8(float* p, v8f v) {
    v4f a, b;
#pragma unroll
    for (int i = 0; i < 4; ++i) { a[i] = v[i]; b[i] = v[4 + i]; }
    *(v4f*)p = a;
    *(v4f*)(p + 4) = b;
}
__device__ __forceinline__ void accum8(float* p, v8f v) {
    v4f a = *(v4f*)p;
    v4f b = *(v4f*)(p + 4);
#pragma unroll
    for (int i = 0; i < 4; ++i) { a[i] += v[i]; b[i] += v[4 + i]; }
    *(v4f*)p = a;
    *(v4f*)(p + 4) = b;
}

// Fetch a broadcast-vector B fragment from a bf16 pair-packed LDS buffer.
__device__ __forceinline__ v8u ldsB(const unsigned int* buf, int ks, int lane) {
    int base = ks * 16 + ((lane & 16) ? 4 : 0);
    v4u lo = *(const v4u*)(buf + base);
    v4u hi = *(const v4u*)(buf + base + 8);
    v8u b;
#pragma unroll
    for (int i = 0; i < 4; ++i) { b[i] = lo[i]; b[4 + i] = hi[i]; }
    return b;
}

// A fragment from a tile base pointer with constant byte offset (ks*1KB),
// letting the compiler fold the walk into the 24-bit instruction offset.
__device__ __forceinline__ v8u ldA(const unsigned int* tile_base, int ks) {
    return *(const v8u*)(tile_base + ks * 256);
}

// ---------------------------------------------------------------------------
// Tensor Data Mover: async 1-D copy of 512 fp32 (2KB) global -> LDS.
// D# per CDNA5 ISA ch.8: count=1, type=2, data_size=4B (code 2),
// tensor_dim0 = tile_dim0 = 512, 1-D tile, no pad/multicast/iterate.
// ---------------------------------------------------------------------------
__device__ __forceinline__ void tdm_load_cond(unsigned int lds_addr,
                                              const float* gptr) {
#if HAVE_TDM
    unsigned long long ga = (unsigned long long)(__UINTPTR_TYPE__)gptr;
    u32x4 g0;
    g0[0] = 1u;                                    // count=1, is_restore=0
    g0[1] = lds_addr;                              // lds_addr[31:0]
    g0[2] = (unsigned int)ga;                      // global_addr[31:0]
    g0[3] = (unsigned int)(ga >> 32) | (2u << 30); // global_addr[56:32]|type=2
    i32x8 g1;
    g1[0] = (int)(2u << 16);       // data_size = 4B
    g1[1] = (int)(512u << 16);     // tensor_dim0[15:0] in [31:16]
    g1[2] = (int)(1u << 16);       // tensor_dim0 hi = 0, tensor_dim1 = 1
    g1[3] = (int)(512u << 16);     // tensor_dim1 hi = 0, tile_dim0 = 512
    g1[4] = 0;                     // tile_dim1 = tile_dim2 = 0 (unused)
    g1[5] = 512;                   // tensor_dim0_stride
    g1[6] = 0;
    g1[7] = 0;
    i32x4 z4; z4[0] = 0; z4[1] = 0; z4[2] = 0; z4[3] = 0;
#if __clang_major__ >= 23
    i32x8 z8;
#pragma unroll
    for (int i = 0; i < 8; ++i) z8[i] = 0;
    __builtin_amdgcn_tensor_load_to_lds(g0, g1, z4, z4, z8, 0);
#else
    __builtin_amdgcn_tensor_load_to_lds(g0, g1, z4, z4, 0);
#endif
#else
    (void)lds_addr; (void)gptr;
#endif
}

__device__ __forceinline__ void tdm_wait() {
#if HAVE_TDM
    __builtin_amdgcn_s_wait_tensorcnt((short)0);
#endif
}

// ---------------------------------------------------------------------------
// Workspace layout (bytes)
// ---------------------------------------------------------------------------
#define OFF_CONDS  0ull                          // 1024*15360 f32 = 62914560
#define OFF_WVCAT  62914560ull                   // 30*512 frags * 1KB
#define OFF_WO     78643200ull                   // 29*256 frags * 1KB
#define OFF_WOL    86245376ull                   // 128 frags
#define OFF_E1     86376448ull
#define OFF_E2     86507520ull
#define OFF_RING   86638592ull                   // 3069 slots * 512B
#define RING_DW    (3069u * 128u)

// ---------------------------------------------------------------------------
// Persistent sequential generator: 1 workgroup, 512 threads = 16 wave32s.
// ---------------------------------------------------------------------------
__global__ void __launch_bounds__(512)
wavenet_generate(const float* __restrict__ emb,
                 const float* __restrict__ wob,      // [29][512]
                 const float* __restrict__ wobl,     // [256]
                 const float* __restrict__ e1b,      // [256]
                 const float* __restrict__ e2b,      // [256]
                 const float* __restrict__ samples,  // [1024]
                 const float* __restrict__ conds,    // [1024][15360]
                 const unsigned int* __restrict__ wvcat,
                 const unsigned int* __restrict__ wo,
                 const unsigned int* __restrict__ wol,
                 const unsigned int* __restrict__ e1w,
                 const unsigned int* __restrict__ e2w,
                 unsigned int* __restrict__ ring,
                 float* __restrict__ out) {
    __shared__ __align__(16) unsigned int s_xcatbf[256]; // [tap|x] packed bf16
    __shared__ __align__(16) unsigned int s_zbf[128];    // gated z packed
    __shared__ __align__(16) unsigned int s_sbf[128];    // relu(skip) packed
    __shared__ __align__(16) unsigned int s_hbf[128];    // relu(hid) packed
    __shared__ __align__(16) float s_cond[2][512];       // TDM double buffer
    __shared__ __align__(16) float s_x[256];
    __shared__ __align__(16) float s_h[512];
    __shared__ __align__(16) float s_skip[256];
    __shared__ __align__(16) float s_red[256];
    __shared__ int   s_ptr[NLAY];
    __shared__ int   s_ringoff[NLAY];
    __shared__ int   s_idx;

    const int tid   = threadIdx.x;
    const int lane  = tid & 31;
    const int wave  = tid >> 5;
    const int rbase = (lane & 16) ? 8 : 0;

    if (tid < NLAY) {
        s_ptr[tid] = 0;
        int off = 0;
        for (int i = 0; i < tid; ++i) off += (1 << (i % 10));
        s_ringoff[tid] = off;
    }
    if (tid < 256) s_x[tid] = emb[127 * 256 + tid];  // emb[NUM_CLS/2 - 1]
    __syncthreads();

    const unsigned int cond_lds0 =
        (unsigned int)(__UINTPTR_TYPE__)&s_cond[0][0];
    const unsigned int cond_lds1 =
        (unsigned int)(__UINTPTR_TYPE__)&s_cond[1][0];

    for (int t = 0; t < TLEN; ++t) {
        if (tid < 256) s_skip[tid] = 0.f;
        if (wave == 0)  // async-prefetch cond for layer 0 of this step
            tdm_load_cond(cond_lds0, conds + (size_t)t * 15360);
        __syncthreads();

        for (int j = 0; j < NLAY; ++j) {
            // --- stage A: tap is already bf16-packed in the ring ------------
            int ptr = s_ptr[j];
            unsigned int* slot = ring + ((size_t)(s_ringoff[j] + ptr)) * 128;
            if (tid < 128) {
                s_xcatbf[tid] = slot[tid];                       // tap
                unsigned int dx = f2bf_pair(s_x[2 * tid], s_x[2 * tid + 1]);
                s_xcatbf[128 + tid] = dx;                        // present x
                slot[tid] = dx;                                  // push x
            }
            if (wave == 0) tdm_wait();  // cond[j] landed in LDS
            __syncthreads();
            if (tid == 0) {
                int d = 1 << (j % 10);
                s_ptr[j] = (ptr + 1 >= d) ? 0 : ptr + 1;
            }
            // overlap: TDM-prefetch cond for layer j+1 during the matvec
            if (wave == 0 && j + 1 < NLAY)
                tdm_load_cond((j & 1) ? cond_lds0 : cond_lds1,
                              conds + (size_t)t * 15360 + (size_t)(j + 1) * 512);

            // --- gated conv: h = cond + [WVp|WVc] @ [tap;x], K=512 ----------
#if HAVE_TDM
            const float* condp = &s_cond[j & 1][0];
#else
            const float* condp = conds + (size_t)t * 15360 + (size_t)j * 512;
#endif
            const int mt0 = wave * 2, mt1 = wave * 2 + 1;
            v8f acc0 = load8(condp + mt0 * 16 + rbase);
            v8f acc1 = load8(condp + mt1 * 16 + rbase);
            // Tile base pointers: fragment walk folds into 24-bit offsets.
            const unsigned int* A0 = wvcat + (size_t)j * 131072 +
                                     (size_t)mt0 * 4096 + (size_t)lane * 8;
            const unsigned int* A1 = A0 + 4096;
            __builtin_prefetch((const void*)(A0 + 131072), 0, 1);
#pragma unroll 4
            for (int ks = 0; ks < 16; ++ks) {
                v8u bb = ldsB(s_xcatbf, ks, lane);
                acc0 = wmma_bf16(ldA(A0, ks), bb, acc0);
                acc1 = wmma_bf16(ldA(A1, ks), bb, acc1);
            }
            if ((lane & 15) == 0) {
                store8(&s_h[mt0 * 16 + rbase], acc0);
                store8(&s_h[mt1 * 16 + rbase], acc1);
            }
            __syncthreads();

            // --- z = tanh(h[:256]) * sigmoid(h[256:]), packed directly ------
            if (tid < 128) {
                float z0 = tanhf(s_h[2 * tid]) *
                           (1.f / (1.f + __expf(-s_h[2 * tid + 256])));
                float z1 = tanhf(s_h[2 * tid + 1]) *
                           (1.f / (1.f + __expf(-s_h[2 * tid + 257])));
                s_zbf[tid] = f2bf_pair(z0, z1);
            }
            __syncthreads();

            // --- output projection: residual + skip -------------------------
            if (j < NLAY - 1) {
                v8f b0 = load8(wob + (size_t)j * 512 + mt0 * 16 + rbase);
                v8f b1 = load8(wob + (size_t)j * 512 + mt1 * 16 + rbase);
                const unsigned int* P0 = wo + (size_t)j * 65536 +
                                         (size_t)mt0 * 2048 + (size_t)lane * 8;
                const unsigned int* P1 = P0 + 2048;
#pragma unroll 4
                for (int ks = 0; ks < 8; ++ks) {
                    v8u bb = ldsB(s_zbf, ks, lane);
                    b0 = wmma_bf16(ldA(P0, ks), bb, b0);
                    b1 = wmma_bf16(ldA(P1, ks), bb, b1);
                }
                // rows 0..255 -> residual, 256..511 -> skip: wave-uniform
                if ((lane & 15) == 0) {
                    if (wave < 8) {
                        accum8(&s_x[wave * 32 + rbase], b0);
                        accum8(&s_x[wave * 32 + 16 + rbase], b1);
                    } else {
                        accum8(&s_skip[wave * 32 - 256 + rbase], b0);
                        accum8(&s_skip[wave * 32 - 240 + rbase], b1);
                    }
                }
            } else {
                // last layer: skip-only projection, M=256, 1 tile per wave
                const int mt = wave;
                v8f a = load8(wobl + mt * 16 + rbase);
                const unsigned int* PL = wol + (size_t)mt * 2048 +
                                         (size_t)lane * 8;
#pragma unroll 4
                for (int ks = 0; ks < 8; ++ks) {
                    v8u bb = ldsB(s_zbf, ks, lane);
                    a = wmma_bf16(ldA(PL, ks), bb, a);
                }
                if ((lane & 15) == 0)
                    accum8(&s_skip[mt * 16 + rbase], a);
            }
            __syncthreads();
        }

        // --- end stack ------------------------------------------------------
        if (tid < 128)
            s_sbf[tid] = f2bf_pair(fmaxf(s_skip[2 * tid], 0.f),
                                   fmaxf(s_skip[2 * tid + 1], 0.f));
        __syncthreads();
        {
            const int mt = wave;
            v8f acc = load8(e1b + mt * 16 + rbase);
            const unsigned int* P1 = e1w + (size_t)mt * 2048 + (size_t)lane * 8;
#pragma unroll 4
            for (int ks = 0; ks < 8; ++ks) {
                v8u bb = ldsB(s_sbf, ks, lane);
                acc = wmma_bf16(ldA(P1, ks), bb, acc);
            }
            // hid = relu(...) written packed for the next matvec
            if ((lane & 15) == 0) {
                v4u hp;
#pragma unroll
                for (int q = 0; q < 4; ++q)
                    hp[q] = f2bf_pair(fmaxf(acc[2 * q], 0.f),
                                      fmaxf(acc[2 * q + 1], 0.f));
                *(v4u*)(s_hbf + ((mt * 16 + rbase) >> 1)) = hp;
            }
            __syncthreads();

            // logits = e2 @ hid + b2
            v8f acc2 = load8(e2b + mt * 16 + rbase);
            const unsigned int* P2 = e2w + (size_t)mt * 2048 + (size_t)lane * 8;
#pragma unroll 4
            for (int ks = 0; ks < 8; ++ks) {
                v8u bb = ldsB(s_hbf, ks, lane);
                acc2 = wmma_bf16(ldA(P2, ks), bb, acc2);
            }
            if ((lane & 15) == 0)
                store8(&s_h[mt * 16 + rbase], acc2);   // logits
            __syncthreads();
        }

        // --- softmax + inverse-CDF sample -----------------------------------
        if (tid < 256) s_red[tid] = s_h[tid];
        __syncthreads();
        for (int s = 128; s > 0; s >>= 1) {
            if (tid < s) s_red[tid] = fmaxf(s_red[tid], s_red[tid + s]);
            __syncthreads();
        }
        float mx = s_red[0];
        __syncthreads();
        if (tid < 256) s_h[tid] = __expf(s_h[tid] - mx);  // probs (unnorm)
        __syncthreads();
        if (tid < 256) s_red[tid] = s_h[tid];
        __syncthreads();
        for (int s = 128; s > 0; s >>= 1) {
            if (tid < s) s_red[tid] += s_red[tid + s];
            __syncthreads();
        }
        if (tid == 0) {
            float inv = 1.f / s_red[0];
            float u = samples[t];
            float cum = 0.f;
            int idx = 0;
            bool found = false;
            for (int i = 0; i < 256; ++i) {
                cum += s_h[i] * inv;
                if (!found && cum > u) { idx = i; found = true; }
            }
            s_idx = idx;
            out[t] = (float)idx;
        }
        __syncthreads();
        int idx = s_idx;
        if (tid < 256) s_x[tid] = emb[(size_t)idx * 256 + tid];
        __syncthreads();
    }
}

// ---------------------------------------------------------------------------
// Weight pre-tiling: [WV_past | WV_present] -> bf16 fragment-major
// ---------------------------------------------------------------------------
__global__ void tile_wvcat(const float* __restrict__ wvp,
                           const float* __restrict__ wvc,
                           unsigned int* __restrict__ dst) {
    const size_t total = (size_t)NLAY * 512 * 512 / 2;  // dwords
    for (size_t dw = (size_t)blockIdx.x * blockDim.x + threadIdx.x;
         dw < total; dw += (size_t)gridDim.x * blockDim.x) {
        int p = (int)(dw & 7);
        int l = (int)((dw >> 3) & 31);
        size_t frag = dw >> 8;
        int ks = (int)(frag & 15);
        int mt = (int)((frag >> 4) & 31);
        int j  = (int)(frag >> 9);
        int m = mt * 16 + (l & 15);
        int k = ks * 32 + kidx(l, p);
        float v0, v1;
        size_t row = ((size_t)j * 512 + m) * 256;
        if (k < 256) { v0 = wvp[row + k];       v1 = wvp[row + k + 1]; }
        else         { v0 = wvc[row + k - 256]; v1 = wvc[row + k - 255]; }
        dst[dw] = f2bf_pair(v0, v1);
    }
}

// Generic M x 256 row-major fp32 -> bf16 fragment-major (frag = mt*8 + ks).
__global__ void tile_mat(const float* __restrict__ src,
                         unsigned int* __restrict__ dst, int M) {
    const size_t total = (size_t)M * 128;  // dwords
    for (size_t dw = (size_t)blockIdx.x * blockDim.x + threadIdx.x;
         dw < total; dw += (size_t)gridDim.x * blockDim.x) {
        int p = (int)(dw & 7);
        int l = (int)((dw >> 3) & 31);
        size_t frag = dw >> 8;
        int ks = (int)(frag & 7);
        int mt = (int)(frag >> 3);
        int m = mt * 16 + (l & 15);
        int k = ks * 32 + kidx(l, p);
        size_t row = (size_t)m * 256;
        dst[dw] = f2bf_pair(src[row + k], src[row + k + 1]);
    }
}

__global__ void zero_ring(unsigned int* __restrict__ ring, unsigned int n) {
    for (unsigned int i = blockIdx.x * blockDim.x + threadIdx.x; i < n;
         i += gridDim.x * blockDim.x)
        ring[i] = 0u;
}

// ---------------------------------------------------------------------------
// Conditioning GEMM: conds[t][m] = sum_k condition_W[m][k] * y[k][t]
// ---------------------------------------------------------------------------
__global__ void __launch_bounds__(256)
cond_gemm(const float* __restrict__ cw, const float* __restrict__ y,
          float* __restrict__ conds) {
    int lane = threadIdx.x & 31;
    int w    = threadIdx.x >> 5;
    int mtile = blockIdx.x * 8 + w;   // 960 tiles of M=15360
    int ttile = blockIdx.y;           // 64 tiles of T=1024
    int rbase = (lane & 16) ? 8 : 0;
    int mrow = mtile * 16 + (lane & 15);
    int tcol = ttile * 16 + (lane & 15);
    v8f acc = vzero8();
#pragma unroll
    for (int ks = 0; ks < 3; ++ks) {
        v8u a, b;
#pragma unroll
        for (int p = 0; p < 8; ++p) {
            int k = ks * 32 + kidx(lane, p);
            float a0 = (k     < CONDC) ? cw[(size_t)mrow * CONDC + k]     : 0.f;
            float a1 = (k + 1 < CONDC) ? cw[(size_t)mrow * CONDC + k + 1] : 0.f;
            float b0 = (k     < CONDC) ? y[(size_t)k * TLEN + tcol]       : 0.f;
            float b1 = (k + 1 < CONDC) ? y[(size_t)(k + 1) * TLEN + tcol] : 0.f;
            a[p] = f2bf_pair(a0, a1);
            b[p] = f2bf_pair(b0, b1);
        }
        acc = wmma_bf16(a, b, acc);
    }
#pragma unroll
    for (int r = 0; r < 8; ++r)
        conds[(size_t)tcol * 15360 + mtile * 16 + rbase + r] = acc[r];
}

// ---------------------------------------------------------------------------
// Launch
// ---------------------------------------------------------------------------
extern "C" void kernel_launch(void* const* d_in, const int* in_sizes, int n_in,
                              void* d_out, int out_size, void* d_ws, size_t ws_size,
                              hipStream_t stream) {
    (void)in_sizes; (void)n_in; (void)out_size; (void)ws_size;
    const float* y       = (const float*)d_in[0];
    const float* emb     = (const float*)d_in[1];
    const float* cw      = (const float*)d_in[2];
    const float* wvp     = (const float*)d_in[3];
    const float* wvc     = (const float*)d_in[4];
    const float* wow     = (const float*)d_in[5];
    const float* wob     = (const float*)d_in[6];
    const float* wowl    = (const float*)d_in[7];
    const float* wobl    = (const float*)d_in[8];
    const float* e1wf    = (const float*)d_in[9];
    const float* e1b     = (const float*)d_in[10];
    const float* e2wf    = (const float*)d_in[11];
    const float* e2b     = (const float*)d_in[12];
    const float* samples = (const float*)d_in[13];

    char* ws = (char*)d_ws;
    float*        conds = (float*)(ws + OFF_CONDS);
    unsigned int* wvcat = (unsigned int*)(ws + OFF_WVCAT);
    unsigned int* wo    = (unsigned int*)(ws + OFF_WO);
    unsigned int* wol   = (unsigned int*)(ws + OFF_WOL);
    unsigned int* e1w   = (unsigned int*)(ws + OFF_E1);
    unsigned int* e2w   = (unsigned int*)(ws + OFF_E2);
    unsigned int* ring  = (unsigned int*)(ws + OFF_RING);
    float* out = (float*)d_out;

    // Weight pre-tiling into WMMA fragment layout (bf16)
    tile_wvcat<<<2048, 256, 0, stream>>>(wvp, wvc, wvcat);
    tile_mat<<<1024, 256, 0, stream>>>(wow,  wo,  29 * 512);
    tile_mat<<<128,  256, 0, stream>>>(wowl, wol, 256);
    tile_mat<<<128,  256, 0, stream>>>(e1wf, e1w, 256);
    tile_mat<<<128,  256, 0, stream>>>(e2wf, e2w, 256);
    zero_ring<<<512, 256, 0, stream>>>(ring, RING_DW);

    // Parallel conditioning GEMM: conds[t][m]
    cond_gemm<<<dim3(120, 64), 256, 0, stream>>>(cw, y, conds);

    // Sequential autoregressive generation (single persistent workgroup)
    wavenet_generate<<<1, 512, 0, stream>>>(emb, wob, wobl, e1b, e2b, samples,
                                            conds, wvcat, wo, wol, e1w, e2w,
                                            ring, out);
}
